// WindowAttention3D_12764642803881
// MI455X (gfx1250) — matrix operand order repaired and checked by reading, not compiled, not run on hardware
//
#include <hip/hip_runtime.h>

// ---------------------------------------------------------------------------
// Fused 3D window attention for MI455X (gfx1250, wave32, WMMA).
// One workgroup per window; qkv / scores / probs / V^T all live in LDS
// (293 KB of the 320 KB WGP LDS). Matrix math via v_wmma_f32_16x16x32_f16.
// Weight GEMMs are register-blocked 2 M-tiles per B-fragment to halve L2
// weight traffic; next weight tile is prefetched (global_prefetch_b8).
// ---------------------------------------------------------------------------

typedef __attribute__((ext_vector_type(16))) _Float16 v16h;
typedef __attribute__((ext_vector_type(8)))  _Float16 v8h;
typedef __attribute__((ext_vector_type(8)))  float    v8f;

#define DIM    192
#define HEADS  6
#define HD     32
#define NTOK   98
#define NPAD   112            // 7 * 16
#define KPAD   128            // attention K padded for 16x16x32
#define TC     576            // 3 * DIM
#define BTOT   2048
#define NWIN   512
#define NN     (NTOK * NTOK)  // 9604

union F8  { v8f  v; float    f[8];  };
union H16 { v16h v; v8h      h[2];  };

#define WMMA_F16(a, b, c) \
    __builtin_amdgcn_wmma_f32_16x16x32_f16(false, (a), false, (b), (short)0, (c), false, false)

// A-matrix 16x32 f16 fragment (ISA 7.12.2): lanes 0-15 row M=lane, K 0-7 &
// 16-23; lanes 16-31 same rows, K 8-15 & 24-31.  Two 16B loads per lane.
__device__ __forceinline__ v16h load_afrag(const _Float16* base, int stride, int lane) {
    int r   = lane & 15;
    int klo = (lane < 16) ? 0 : 8;
    H16 u;
    u.h[0] = *(const v8h*)(base + r * stride + klo);
    u.h[1] = *(const v8h*)(base + r * stride + klo + 16);
    return u.v;
}

// B-matrix 32x16 f16 fragment: lane's column = lane&15; lanes 0-15 hold
// K 0-7 & 16-23, lanes 16-31 hold K 8-15 & 24-31.  Two 16B loads per lane.
__device__ __forceinline__ v16h load_bfrag(const _Float16* base, int stride, int lane) {
    int c   = lane & 15;
    int klo = (lane < 16) ? 0 : 8;
    H16 u;
    u.h[0] = *(const v8h*)(base + c * stride + klo);
    u.h[1] = *(const v8h*)(base + c * stride + klo + 16);
    return u.v;
}

// Store one accumulator tile into qkv LDS (f16), mirroring V columns into V^T.
__device__ __forceinline__ void store_qkv_tile(const F8& acc, int mt, int col, float bc,
                                               _Float16* qkvs, _Float16* vt, int lane) {
#pragma unroll
    for (int i = 0; i < 8; ++i) {
        int row = mt * 16 + ((lane < 16) ? i : i + 8);
        _Float16 hv = (_Float16)(acc.f[i] + bc);
        qkvs[row * TC + col] = hv;
        if (col >= 2 * DIM)
            vt[(col - 2 * DIM) * KPAD + row] = hv;
    }
}

// ---------------------------------------------------------------------------
// Prep: f32->f16 weight conversion + relative-position-bias gather.
// ---------------------------------------------------------------------------
__global__ void prep_kernel(const float* __restrict__ qkv_w,
                            const float* __restrict__ proj_w,
                            const float* __restrict__ bias_table,
                            const int*   __restrict__ rel_idx,
                            _Float16* __restrict__ wq,
                            _Float16* __restrict__ wp,
                            float*    __restrict__ biasf) {
    int tid  = blockIdx.x * blockDim.x + threadIdx.x;
    int nthr = gridDim.x * blockDim.x;
    for (int i = tid; i < TC * DIM; i += nthr)  wq[i] = (_Float16)qkv_w[i];
    for (int i = tid; i < DIM * DIM; i += nthr) wp[i] = (_Float16)proj_w[i];
    for (int i = tid; i < NN; i += nthr) {
        int t = rel_idx[i];
#pragma unroll
        for (int h = 0; h < HEADS; ++h)
            biasf[h * NN + i] = bias_table[t * HEADS + h];
    }
}

// ---------------------------------------------------------------------------
// Main fused kernel: one window per workgroup, 8 wave32 per workgroup.
// ---------------------------------------------------------------------------
__global__ void __launch_bounds__(256, 1)
winattn_kernel(const float* __restrict__ x,
               const float* __restrict__ mask,
               const float* __restrict__ qkv_b,
               const float* __restrict__ proj_b,
               const _Float16* __restrict__ wq,
               const _Float16* __restrict__ wp,
               const float* __restrict__ biasf,
               float* __restrict__ out) {
    extern __shared__ char smem_raw[];
    _Float16* xs   = (_Float16*)smem_raw;               // [NPAD][DIM]  x, reused as attn-out
    _Float16* qkvs = xs + NPAD * DIM;                   // [NPAD][TC]
    float*    sbuf = (float*)(qkvs + NPAD * TC);        // [NPAD][NPAD] scores
    _Float16* pbuf = (_Float16*)(sbuf + NPAD * NPAD);   // [NPAD][KPAD] probs
    _Float16* vt   = pbuf + NPAD * KPAD;                // [DIM][KPAD]  V^T (k-major)

    const int tid  = threadIdx.x;
    const int lane = tid & 31;
    const int wave = tid >> 5;
    const int b    = blockIdx.x;
    const float scale = 0.17677669529663687f;           // 32^-0.5

    // ---- Phase 0: stage x -> LDS as f16, zero padding rows; zero V^T ------
    const float* xg = x + (size_t)b * NTOK * DIM;
    for (int i = tid; i < NPAD * DIM; i += 256) {
        int r = i / DIM;
        xs[i] = (r < NTOK) ? (_Float16)xg[i] : (_Float16)0.f;
    }
    {   // zero V^T once (covers the token-padding columns 112..127)
        int* vz = (int*)vt;
        for (int i = tid; i < (DIM * KPAD) / 2; i += 256) vz[i] = 0;
    }
    __syncthreads();

    // ---- Phase 1: qkv = x @ Wqkv^T + b  (M=112, N=576, K=192) -------------
    // Register-blocked: 2 M-tiles share each B fragment (4 M-pairs x 36 N).
    for (int t = wave; t < 4 * 36; t += 8) {
        int pm = t / 36, nt = t % 36;
        int mt0 = pm * 2, mt1 = mt0 + 1;               // mt1==7 invalid (pm==3)
        F8 acc0 = {}, acc1 = {};
#pragma unroll
        for (int ks = 0; ks < 6; ++ks) {
            v16h bb = load_bfrag(wq + (nt * 16) * DIM + ks * 32, DIM, lane);
            v16h a0 = load_afrag(xs + mt0 * 16 * DIM + ks * 32, DIM, lane);
            acc0.v = WMMA_F16(a0, bb, acc0.v);
            if (mt1 < 7) {                             // wave-uniform
                v16h a1 = load_afrag(xs + mt1 * 16 * DIM + ks * 32, DIM, lane);
                acc1.v = WMMA_F16(a1, bb, acc1.v);
            }
        }
        if (t + 8 < 4 * 36) {                          // prefetch next B tile
            int nt2 = (t + 8) % 36;
            __builtin_prefetch(wq + (nt2 * 16 + (lane & 15)) * DIM +
                               ((lane < 16) ? 0 : 16), 0, 2);
        }
        int col  = nt * 16 + (lane & 15);
        float bc = qkv_b[col];
        store_qkv_tile(acc0, mt0, col, bc, qkvs, vt, lane);
        if (mt1 < 7) store_qkv_tile(acc1, mt1, col, bc, qkvs, vt, lane);
    }
    __syncthreads();

    // ---- Phase 2: per-head attention --------------------------------------
    const float* mrow = mask + (size_t)(b % NWIN) * NN;
    for (int h = 0; h < HEADS; ++h) {
        const float* bh = biasf + h * NN;

        // S = scale * q k^T + bias + mask    (49 16x16 tiles, K=32)
        for (int t = wave; t < 49; t += 8) {
            int mt = t / 7, nt = t % 7;
            v16h a  = load_afrag(qkvs + mt * 16 * TC + h * HD, TC, lane);
            v16h bb = load_bfrag(qkvs + nt * 16 * TC + DIM + h * HD, TC, lane);
            F8 acc = {};
            acc.v = WMMA_F16(a, bb, acc.v);
            int n = nt * 16 + (lane & 15);
#pragma unroll
            for (int i = 0; i < 8; ++i) {
                int m   = mt * 16 + ((lane < 16) ? i : i + 8);
                float v = acc.f[i] * scale;
                if (n < NTOK) {
                    if (m < NTOK) v += bh[m * NTOK + n] + mrow[m * NTOK + n];
                } else {
                    v = -3.0e38f;                      // padded key -> prob 0
                }
                sbuf[m * NPAD + n] = v;
            }
        }
        __syncthreads();

        // row softmax -> f16 probs (one thread per row)
        for (int r = tid; r < NPAD; r += 256) {
            float* sr = sbuf + r * NPAD;
            float mx = -3.0e38f;
            for (int c = 0; c < NPAD; ++c) mx = fmaxf(mx, sr[c]);
            float sum = 0.f;
            for (int c = 0; c < NPAD; ++c) {
                float e = __expf(sr[c] - mx);
                sr[c] = e; sum += e;
            }
            float inv = 1.f / sum;
            _Float16* pr = pbuf + r * KPAD;
            for (int c = 0; c < NPAD; ++c)    pr[c] = (_Float16)(sr[c] * inv);
            for (int c = NPAD; c < KPAD; ++c) pr[c] = (_Float16)0.f;
        }
        __syncthreads();

        // O_h = P @ V_h  (7 M-tiles; one A-frag feeds both 16-wide N halves)
        for (int t = wave; t < 7; t += 8) {
            int mt = t;
            F8 acc0 = {}, acc1 = {};
#pragma unroll
            for (int ks = 0; ks < 4; ++ks) {
                v16h a  = load_afrag(pbuf + mt * 16 * KPAD + ks * 32, KPAD, lane);
                v16h b0 = load_bfrag(vt + (h * HD) * KPAD + ks * 32, KPAD, lane);
                v16h b1 = load_bfrag(vt + (h * HD + 16) * KPAD + ks * 32, KPAD, lane);
                acc0.v = WMMA_F16(a, b0, acc0.v);
                acc1.v = WMMA_F16(a, b1, acc1.v);
            }
            int c0 = h * HD + (lane & 15);
#pragma unroll
            for (int i = 0; i < 8; ++i) {
                int m = mt * 16 + ((lane < 16) ? i : i + 8);
                xs[m * DIM + c0]      = (_Float16)acc0.f[i];   // xs reused as O
                xs[m * DIM + c0 + 16] = (_Float16)acc1.f[i];
            }
        }
        __syncthreads();
    }

    // ---- Phase 3: out = O @ Wp^T + b  (M=112, N=192, K=192) ---------------
    for (int t = wave; t < 4 * 12; t += 8) {
        int pm = t / 12, nt = t % 12;
        int mt0 = pm * 2, mt1 = mt0 + 1;
        F8 acc0 = {}, acc1 = {};
#pragma unroll
        for (int ks = 0; ks < 6; ++ks) {
            v16h bb = load_bfrag(wp + (nt * 16) * DIM + ks * 32, DIM, lane);
            v16h a0 = load_afrag(xs + mt0 * 16 * DIM + ks * 32, DIM, lane);
            acc0.v = WMMA_F16(a0, bb, acc0.v);
            if (mt1 < 7) {
                v16h a1 = load_afrag(xs + mt1 * 16 * DIM + ks * 32, DIM, lane);
                acc1.v = WMMA_F16(a1, bb, acc1.v);
            }
        }
        if (t + 8 < 4 * 12) {                          // prefetch next B tile
            int nt2 = (t + 8) % 12;
            __builtin_prefetch(wp + (nt2 * 16 + (lane & 15)) * DIM +
                               ((lane < 16) ? 0 : 16), 0, 2);
        }
        int col  = nt * 16 + (lane & 15);
        float bc = proj_b[col];
        float* og = out + (size_t)b * NTOK * DIM;
#pragma unroll
        for (int i = 0; i < 8; ++i) {
            int m = mt0 * 16 + ((lane < 16) ? i : i + 8);
            if (m < NTOK) og[m * DIM + col] = acc0.f[i] + bc;
        }
        if (mt1 < 7) {
#pragma unroll
            for (int i = 0; i < 8; ++i) {
                int m = mt1 * 16 + ((lane < 16) ? i : i + 8);
                if (m < NTOK) og[m * DIM + col] = acc1.f[i] + bc;
            }
        }
    }
}

// ---------------------------------------------------------------------------
extern "C" void kernel_launch(void* const* d_in, const int* in_sizes, int n_in,
                              void* d_out, int out_size, void* d_ws, size_t ws_size,
                              hipStream_t stream) {
    const float* x      = (const float*)d_in[0];
    const float* mask   = (const float*)d_in[1];
    const float* qkv_w  = (const float*)d_in[2];
    const float* qkv_b  = (const float*)d_in[3];
    const float* proj_w = (const float*)d_in[4];
    const float* proj_b = (const float*)d_in[5];
    const float* btab   = (const float*)d_in[6];
    const int*   ridx   = (const int*)d_in[7];
    float* out = (float*)d_out;

    // workspace: f16 weights + pre-gathered bias (~525 KB)
    _Float16* wq   = (_Float16*)d_ws;              // [576*192] f16
    _Float16* wp   = wq + TC * DIM;                // [192*192] f16
    float*    bsf  = (float*)(wp + DIM * DIM);     // [6*9604]  f32

    prep_kernel<<<256, 256, 0, stream>>>(qkv_w, proj_w, btab, ridx, wq, wp, bsf);

    size_t smem = (size_t)(NPAD * DIM) * 2      // xs / O
                + (size_t)(NPAD * TC) * 2       // qkv
                + (size_t)(NPAD * NPAD) * 4     // scores f32
                + (size_t)(NPAD * KPAD) * 2     // probs f16
                + (size_t)(DIM * KPAD) * 2;     // V^T f16   -> 300,032 B (< 320 KB WGP LDS)

    winattn_kernel<<<BTOT, 256, smem, stream>>>(x, mask, qkv_b, proj_b,
                                                wq, wp, bsf, out);
}
